// AttentionHead_23381801959780
// MI455X (gfx1250) — matrix operand order repaired and checked by reading, not compiled, not run on hardware
//
#include <hip/hip_runtime.h>
#include <hip/hip_bf16.h>

typedef __attribute__((ext_vector_type(16))) __bf16 v16bf;
typedef __attribute__((ext_vector_type(8)))  __bf16 v8bf;
typedef __attribute__((ext_vector_type(8)))  float  v8f;
typedef __attribute__((ext_vector_type(4)))  int    i4;

#define CAT16(lo,hi) __builtin_shufflevector((lo),(hi),0,1,2,3,4,5,6,7,8,9,10,11,12,13,14,15)
#define WMMA_BF16(A,B,C) __builtin_amdgcn_wmma_f32_16x16x32_bf16(false,(A),false,(B),(short)0,(C),false,false)

// CDNA5 async global->LDS DMA (ASYNCcnt-tracked), guarded for toolchain drift.
#if defined(__has_builtin)
#  if __has_builtin(__builtin_amdgcn_global_load_async_to_lds_b128) && \
      __has_builtin(__builtin_amdgcn_s_wait_asynccnt)
#    define USE_ASYNC_LDS 1
#  endif
#endif
#ifndef USE_ASYNC_LDS
#  define USE_ASYNC_LDS 0
#endif

#define AS1 __attribute__((address_space(1)))
#define AS3 __attribute__((address_space(3)))

static constexpr int S_ = 4096;
static constexpr int D_ = 1024;
static constexpr int H_ = 64;
static constexpr float LOG2E = 1.4426950408889634f;

__device__ __forceinline__ float hmax16(float x) {
#pragma unroll
  for (int off = 1; off < 16; off <<= 1) x = fmaxf(x, __shfl_xor(x, off, 32));
  return x;
}
__device__ __forceinline__ float hsum16(float x) {
#pragma unroll
  for (int off = 1; off < 16; off <<= 1) x += __shfl_xor(x, off, 32);
  return x;
}

// ---------------------------------------------------------------------------
// Kernel 0: W [D,H] fp32 -> W^T [H,D] bf16  (3 projections).
// Wq is pre-scaled by 1/sqrt(H) so attention scores need no extra multiply.
// ---------------------------------------------------------------------------
__global__ void __launch_bounds__(256) wconv_kernel(
    const float* __restrict__ Wq, const float* __restrict__ Wk,
    const float* __restrict__ Wv, __bf16* __restrict__ wT)
{
  const int i   = blockIdx.x * 256 + threadIdx.x;   // 0 .. 3*64*1024-1
  const int p   = i >> 16;
  const int rem = i & 65535;
  const int d   = rem >> 6;
  const int h   = rem & 63;
  const float* W = (p == 0) ? Wq : ((p == 1) ? Wk : Wv);
  float w = W[(size_t)d * 64 + h];
  if (p == 0) w *= 0.125f;                          // 1/sqrt(64) folded into Wq
  wT[(size_t)p * 65536 + (size_t)h * 1024 + d] = (__bf16)w;
}

// ---------------------------------------------------------------------------
// Kernel 1: projections via bf16 WMMA, fp32 accumulate.
// blockIdx.y = projection (scalar), one wave per 16-row tile.
// q,k stored [row][h] bf16;  v stored transposed [b][h][s] bf16.
// ---------------------------------------------------------------------------
__global__ void __launch_bounds__(256) proj_kernel(
    const float* __restrict__ Q, const float* __restrict__ K,
    const float* __restrict__ V, const __bf16* __restrict__ wT,
    __bf16* __restrict__ qb, __bf16* __restrict__ kb, __bf16* __restrict__ vT)
{
  const int t    = blockIdx.x * 8 + (threadIdx.x >> 5);   // row tile 0..1023
  const int p    = blockIdx.y;                            // projection (SGPR)
  const int lane = threadIdx.x & 31;
  const int m    = lane & 15;
  const int hf   = lane >> 4;
  const float*  X  = (p == 0) ? Q : ((p == 1) ? K : V);
  const __bf16* wp = wT + (size_t)p * (H_ * D_);
  const int base = t * 16;
  // per-lane A stream: row (base+m), groups at +hf*8 and +16+hf*8
  const float* xp = X + (size_t)(base + m) * D_ + hf * 8;

  const v8f VZ = {0.f,0.f,0.f,0.f,0.f,0.f,0.f,0.f};
  v8f acc[4];
#pragma unroll
  for (int nt = 0; nt < 4; ++nt) acc[nt] = VZ;

  // software-pipelined A tile (prefetch next K-chunk before this chunk's WMMAs)
  float4 a0 = *(const float4*)(xp);
  float4 a1 = *(const float4*)(xp + 4);
  float4 a2 = *(const float4*)(xp + 16);
  float4 a3 = *(const float4*)(xp + 20);

#pragma unroll 2
  for (int c = 0; c < D_; c += 32) {
    v16bf a;
    a[0]=(__bf16)a0.x; a[1]=(__bf16)a0.y; a[2]=(__bf16)a0.z; a[3]=(__bf16)a0.w;
    a[4]=(__bf16)a1.x; a[5]=(__bf16)a1.y; a[6]=(__bf16)a1.z; a[7]=(__bf16)a1.w;
    a[8]=(__bf16)a2.x; a[9]=(__bf16)a2.y; a[10]=(__bf16)a2.z; a[11]=(__bf16)a2.w;
    a[12]=(__bf16)a3.x; a[13]=(__bf16)a3.y; a[14]=(__bf16)a3.z; a[15]=(__bf16)a3.w;
    if (c + 32 < D_) {
      const float* nx = xp + c + 32;
      a0 = *(const float4*)(nx);
      a1 = *(const float4*)(nx + 4);
      a2 = *(const float4*)(nx + 16);
      a3 = *(const float4*)(nx + 20);
    }
    // all four B operands up front (one load clause), then four WMMAs
    v16bf bop[4];
#pragma unroll
    for (int nt = 0; nt < 4; ++nt)
      bop[nt] = CAT16(*(const v8bf*)(wp + (size_t)(nt * 16 + m) * D_ + c + 8 * hf), *(const v8bf*)(wp + (size_t)(nt * 16 + m) * D_ + c + 16 + 8 * hf));
#pragma unroll
    for (int nt = 0; nt < 4; ++nt)
      acc[nt] = WMMA_BF16(a, bop[nt], acc[nt]);
  }

  if (p < 2) {
    __bf16* dst = (p == 0) ? qb : kb;
#pragma unroll
    for (int nt = 0; nt < 4; ++nt)
#pragma unroll
      for (int r = 0; r < 8; ++r)
        dst[(size_t)(base + r + 8 * hf) * H_ + nt * 16 + m] = (__bf16)acc[nt][r];
  } else {
#pragma unroll
    for (int nt = 0; nt < 4; ++nt)
#pragma unroll
      for (int r = 0; r < 8; ++r) {
        const int row = base + r + 8 * hf;
        const int bb = row >> 12, s = row & (S_ - 1);
        vT[(size_t)bb * (H_ * S_) + (size_t)(nt * 16 + m) * S_ + s] = (__bf16)acc[nt][r];
      }
  }
}

// ---------------------------------------------------------------------------
// Kernel 2: causal flash attention. Block = 8 waves = 128 queries of one batch.
// K/V chunks (32 keys) staged into double-buffered LDS via async global->LDS
// DMA (ASYNCcnt), overlapped with WMMA compute of the previous chunk.
// ---------------------------------------------------------------------------
#define KSTR 144   // bytes/row: 64 bf16 (=128B) + 16B pad, odd multiple of 16
#define VSTR 80    // bytes/row: 32 bf16 (=64B)  + 16B pad
#define PSTR 80

__global__ void __launch_bounds__(256) attn_kernel(
    const __bf16* __restrict__ qb, const __bf16* __restrict__ kb,
    const __bf16* __restrict__ vT, float* __restrict__ out)
{
  constexpr int KB = 32 * KSTR;          // 4608 B per K buffer
  constexpr int VB = 64 * VSTR;          // 5120 B per V buffer
  __shared__ alignas(16) unsigned char smem[2 * KB + 2 * VB + 8 * 16 * PSTR];

  const int b    = blockIdx.y;
  const int qb0  = blockIdx.x * 128;
  const int tid  = threadIdx.x;
  const int warp = tid >> 5;
  const int lane = tid & 31;
  const int m    = lane & 15;
  const int hf   = lane >> 4;
  const int q0   = qb0 + warp * 16;

  // Q tile as the two K(=h)-chunk WMMA A operands (Q already carries 1/sqrt(H))
  const __bf16* qrow = qb + (size_t)(b * S_ + q0 + m) * H_;
  v16bf qa[2];
#pragma unroll
  for (int ci = 0; ci < 2; ++ci) {
    const int c = ci * 32;
    v8bf lo = *(const v8bf*)(qrow + c + hf * 16);
    v8bf hi = *(const v8bf*)(qrow + c + hf * 16 + 8);
    qa[ci] = CAT16(lo, hi);
  }

  const v8f VZ = {0.f,0.f,0.f,0.f,0.f,0.f,0.f,0.f};
  v8f o[4];
#pragma unroll
  for (int nt = 0; nt < 4; ++nt) o[nt] = VZ;
  float mrun[8], lrun[8];
#pragma unroll
  for (int r = 0; r < 8; ++r) { mrun[r] = -__builtin_inff(); lrun[r] = 0.f; }

  __bf16* pmy = (__bf16*)(smem + 2 * KB + 2 * VB + warp * 16 * PSTR);
  const int nchunks = blockIdx.x * 4 + 4;   // keys 0 .. qb0+127 in 32-wide chunks

  // cooperative stage of one 32-key chunk: K 32x64 bf16, V^T 64x32 bf16
  auto stage = [&](int buf, int kb0) {
    unsigned char* kdst = smem + buf * KB;
    unsigned char* vdst = smem + 2 * KB + buf * VB;
    const int kr = tid >> 3, ks = tid & 7;
    const int vr = tid >> 2, vs = tid & 3;
    const __bf16* kg = kb + (size_t)(b * S_ + kb0 + kr) * H_ + ks * 8;
    const __bf16* vg = vT + (size_t)b * (H_ * S_) + (size_t)vr * S_ + kb0 + vs * 8;
#if USE_ASYNC_LDS
    __builtin_amdgcn_global_load_async_to_lds_b128(
        (AS1 i4*)kg, (AS3 i4*)(kdst + kr * KSTR + ks * 16), 0, 0);
    __builtin_amdgcn_global_load_async_to_lds_b128(
        (AS1 i4*)vg, (AS3 i4*)(vdst + vr * VSTR + vs * 16), 0, 0);
#else
    *(v8bf*)(kdst + kr * KSTR + ks * 16) = *(const v8bf*)kg;
    *(v8bf*)(vdst + vr * VSTR + vs * 16) = *(const v8bf*)vg;
#endif
  };

  auto compute = [&](int buf, int kb0) {
    const unsigned char* ksh = smem + buf * KB;
    const unsigned char* vsh = smem + 2 * KB + buf * VB;
    // S = Q K^T  (two 16x16 key tiles, K-dim = 64 -> 2 WMMAs each)
    v8f sacc[2];
#pragma unroll
    for (int kt = 0; kt < 2; ++kt) {
      v8f z = VZ;
#pragma unroll
      for (int ci = 0; ci < 2; ++ci) {
        const int c = ci * 32;
        const __bf16* kr = (const __bf16*)(ksh + (16 * kt + m) * KSTR) + c + 16 * hf;
        v8bf lo = *(const v8bf*)(kr);
        v8bf hi = *(const v8bf*)(kr + 8);
        z = WMMA_BF16(qa[ci], CAT16(lo, hi), z);
      }
      sacc[kt] = z;
    }
    // online softmax over the 32-key chunk; mask only needed near the diagonal
    const bool needmask = (kb0 + 31 > q0);     // wave-uniform
    float pv0[8], pv1[8];
#pragma unroll
    for (int r = 0; r < 8; ++r) {
      float s0 = sacc[0][r];
      float s1 = sacc[1][r];
      if (needmask) {
        const int qi = q0 + r + 8 * hf;
        if (kb0 + m > qi)      s0 = -__builtin_inff();
        if (kb0 + 16 + m > qi) s1 = -__builtin_inff();
      }
      const float tmax = hmax16(fmaxf(s0, s1));
      const float mnew = fmaxf(mrun[r], tmax);
      const float resc = exp2f((mrun[r] - mnew) * LOG2E);
      const float p0 = exp2f((s0 - mnew) * LOG2E);
      const float p1 = exp2f((s1 - mnew) * LOG2E);
      lrun[r] = lrun[r] * resc + hsum16(p0 + p1);
      mrun[r] = mnew;
#pragma unroll
      for (int nt = 0; nt < 4; ++nt) o[nt][r] *= resc;
      pv0[r] = p0; pv1[r] = p1;
    }
    // P: C layout -> LDS row-major 16x32 -> reread as A operand
#pragma unroll
    for (int r = 0; r < 8; ++r) {
      pmy[(r + 8 * hf) * (PSTR / 2) + m]      = (__bf16)pv0[r];
      pmy[(r + 8 * hf) * (PSTR / 2) + 16 + m] = (__bf16)pv1[r];
    }
    const __bf16* pr = pmy + m * (PSTR / 2);
    v8bf plo = *(const v8bf*)(pr + hf * 16);
    v8bf phi = *(const v8bf*)(pr + hf * 16 + 8);
    const v16bf pa = CAT16(plo, phi);
    // O += P V   (B operand = rows of V^T, 16 contiguous keys per lane)
#pragma unroll
    for (int nt = 0; nt < 4; ++nt) {
      const __bf16* vr = (const __bf16*)(vsh + (nt * 16 + m) * VSTR) + 16 * hf;
      v8bf blo = *(const v8bf*)(vr);
      v8bf bhi = *(const v8bf*)(vr + 8);
      o[nt] = WMMA_BF16(pa, CAT16(blo, bhi), o[nt]);
    }
  };

#if USE_ASYNC_LDS
  stage(0, 0);                                      // prime chunk 0
  for (int j = 0; j < nchunks; ++j) {
    __builtin_amdgcn_s_wait_asynccnt(0);            // own DMAs done...
    __syncthreads();                                // ...everyone's DMAs + readers done
    const int cur = j & 1;
    if (j + 1 < nchunks)                            // prefetch next chunk into other buffer
      stage(cur ^ 1, (j + 1) * 32);
    if (j * 32 <= q0 + 15)                          // wave-uniform causal filter
      compute(cur, j * 32);
  }
#else
  for (int j = 0; j < nchunks; ++j) {
    stage(0, j * 32);
    __syncthreads();
    if (j * 32 <= q0 + 15)
      compute(0, j * 32);
    __syncthreads();
  }
#endif

  const size_t obase = (size_t)(b * S_ + q0) * H_;
#pragma unroll
  for (int r = 0; r < 8; ++r) {
    const float inv = 1.0f / lrun[r];
    const int row = r + 8 * hf;
#pragma unroll
    for (int nt = 0; nt < 4; ++nt)
      out[obase + (size_t)row * H_ + nt * 16 + m] = o[nt][r] * inv;
  }
}

// ---------------------------------------------------------------------------
extern "C" void kernel_launch(void* const* d_in, const int* in_sizes, int n_in,
                              void* d_out, int out_size, void* d_ws, size_t ws_size,
                              hipStream_t stream)
{
  (void)in_sizes; (void)n_in; (void)out_size; (void)ws_size;
  const float* Q  = (const float*)d_in[0];
  const float* K  = (const float*)d_in[1];
  const float* V  = (const float*)d_in[2];
  const float* Wq = (const float*)d_in[3];
  const float* Wk = (const float*)d_in[4];
  const float* Wv = (const float*)d_in[5];
  float* out = (float*)d_out;

  char* ws = (char*)d_ws;
  __bf16* wT   = (__bf16*)(ws);                              // 384 KB
  __bf16* qbuf = (__bf16*)(ws + 393216);                     // 2 MB
  __bf16* kbuf = (__bf16*)(ws + 393216 + 2097152);           // 2 MB
  __bf16* vTb  = (__bf16*)(ws + 393216 + 2 * 2097152);       // 2 MB (transposed)

  wconv_kernel<<<768, 256, 0, stream>>>(Wq, Wk, Wv, wT);
  proj_kernel<<<dim3(128, 3), 256, 0, stream>>>(Q, K, V, wT, qbuf, kbuf, vTb);
  attn_kernel<<<dim3(32, 4), 256, 0, stream>>>(qbuf, kbuf, vTb, out);
}
